// ALiBiAttention_48576080118371
// MI455X (gfx1250) — matrix operand order repaired and checked by reading, not compiled, not run on hardware
//
#include <hip/hip_runtime.h>

#define DEV __device__ __forceinline__

typedef __attribute__((ext_vector_type(16))) _Float16 v16h;
typedef __attribute__((ext_vector_type(8)))  _Float16 v8h;
typedef __attribute__((ext_vector_type(4)))  _Float16 v4h;
typedef __attribute__((ext_vector_type(8)))  float    v8f;
typedef __attribute__((ext_vector_type(4)))  unsigned int u32x4;
typedef __attribute__((ext_vector_type(4)))  int      i32x4;
typedef __attribute__((ext_vector_type(8)))  int      i32x8;

static constexpr int D_MODEL   = 1024;
static constexpr int NUM_HEADS = 16;
static constexpr int HEAD_DIM  = 64;
static constexpr int BB        = 4;
static constexpr int TT        = 2048;
static constexpr int MROWS     = BB * TT;   // 8192

// LDS row strides with TDM padding (4 DWORDs pad per row)
static constexpr int KROW = 72;   // 64 data halfs + 8 pad halfs (144 B)
static constexpr int VROW = 40;   // 32 data halfs + 8 pad halfs (80 B)

#if defined(__HIP_DEVICE_COMPILE__) && \
    __has_builtin(__builtin_amdgcn_tensor_load_to_lds) && \
    __has_builtin(__builtin_amdgcn_s_wait_tensorcnt)
#define HAVE_TDM 1
#else
#define HAVE_TDM 0
#endif

DEV v8h ld8(const _Float16* p) { return *reinterpret_cast<const v8h*>(p); }
DEV v16h cat(v8h a, v8h b) {
  return __builtin_shufflevector(a, b, 0,1,2,3,4,5,6,7,8,9,10,11,12,13,14,15);
}
DEV v8f wmma_f16(v16h a, v16h b, v8f c) {
  // D = A(16x32 f16) * B(32x16 f16) + C(16x16 f32)
  return __builtin_amdgcn_wmma_f32_16x16x32_f16(false, a, false, b, (short)0, c, false, false);
}

#if HAVE_TDM
// D# group1 word0: data_size=2B | pad_enable | pad_interval | pad_amount(=4 DW)
static constexpr unsigned TDM_W0_K =
    (1u << 16) | (1u << 20) | (4u << 22) | (3u << 25);  // pad every 32 DW
static constexpr unsigned TDM_W0_V =
    (1u << 16) | (1u << 20) | (3u << 22) | (3u << 25);  // pad every 16 DW

// 2D tile load: tile0 x tile1 elements (2B each) from row-major tensor with
// row stride stride0 (elements), packed (plus pad) into LDS at lds_off.
DEV void tdm_load_2d(unsigned lds_off, const _Float16* gptr, unsigned w0,
                     unsigned dim0, unsigned tile0, unsigned tile1,
                     unsigned stride0) {
  unsigned long long ga = (unsigned long long)(uintptr_t)gptr;
  u32x4 g0;
  g0[0] = 1u;                                                  // count=1
  g0[1] = lds_off;                                             // LDS byte addr
  g0[2] = (unsigned)ga;                                        // gaddr[31:0]
  g0[3] = ((unsigned)(ga >> 32) & 0x01FFFFFFu) | (2u << 30);   // [56:32]|type=2
  i32x8 g1;
  g1[0] = (int)w0;                           // mask=0, data_size, pad cfg
  g1[1] = (int)(dim0 << 16);                 // tensor_dim0[15:0]
  g1[2] = (int)(dim0 >> 16);                 // tensor_dim0[31:16], td1 lo=0
  g1[3] = (int)((1u << 4) | (tile0 << 16));  // td1 hi (td1=1<<20) | tile_dim0
  g1[4] = (int)tile1;                        // tile_dim1, tile_dim2=0
  g1[5] = (int)stride0;                      // dim0 stride lo32
  g1[6] = 0;
  g1[7] = 0;
  i32x4 z4 = {0, 0, 0, 0};
#if __clang_major__ >= 23
  i32x8 z8 = {0, 0, 0, 0, 0, 0, 0, 0};
  __builtin_amdgcn_tensor_load_to_lds(g0, g1, z4, z4, z8, 0);
#else
  __builtin_amdgcn_tensor_load_to_lds(g0, g1, z4, z4, 0);
#endif
}
#endif  // HAVE_TDM

// ---------------------------------------------------------------- f32 -> f16
__global__ void cvt_f32_to_f16(const float* __restrict__ in,
                               _Float16* __restrict__ out, int n) {
  int i = (blockIdx.x * blockDim.x + threadIdx.x) * 4;
  if (i < n) {
    float4 v = *reinterpret_cast<const float4*>(in + i);
    v4h h;
    h[0] = (_Float16)v.x; h[1] = (_Float16)v.y;
    h[2] = (_Float16)v.z; h[3] = (_Float16)v.w;
    *reinterpret_cast<v4h*>(out + i) = h;
  }
}

// ------------------------------------------------------------- GEMM core
// Per-wave M32 x N64 x K1024, software-pipelined fragment loads with a
// uniform trip count of 32: the "next" index wraps to 0 on the last step
// (one redundant in-bounds load set) so there is no remainder iteration and
// unroll-by-2 halves schedule symmetrically (no acc copies / hazard nops).
DEV void gemm_core(const _Float16* __restrict__ ap0,
                   const _Float16* __restrict__ ap1,
                   const _Float16* __restrict__ bp,
                   v8f acc[8]) {
  constexpr int K = D_MODEL;
  constexpr size_t BS = (size_t)16 * K;

  v16h a0 = cat(ld8(ap0), ld8(ap0 + 8));
  v16h a1 = cat(ld8(ap1), ld8(ap1 + 8));
  v16h b0 = cat(ld8(bp),          ld8(bp + 8));
  v16h b1 = cat(ld8(bp + BS),     ld8(bp + BS + 8));
  v16h b2 = cat(ld8(bp + 2 * BS), ld8(bp + 2 * BS + 8));
  v16h b3 = cat(ld8(bp + 3 * BS), ld8(bp + 3 * BS + 8));

  for (int k = 0; k < K; k += 32) {
    const int kn = (k + 32) & (K - 1);   // wraps to 0 on last step
    v16h na0 = cat(ld8(ap0 + kn), ld8(ap0 + kn + 8));
    v16h na1 = cat(ld8(ap1 + kn), ld8(ap1 + kn + 8));
    v16h nb0 = cat(ld8(bp + kn),          ld8(bp + kn + 8));
    v16h nb1 = cat(ld8(bp + BS + kn),     ld8(bp + BS + kn + 8));
    v16h nb2 = cat(ld8(bp + 2 * BS + kn), ld8(bp + 2 * BS + kn + 8));
    v16h nb3 = cat(ld8(bp + 3 * BS + kn), ld8(bp + 3 * BS + kn + 8));
    __builtin_prefetch(ap0 + kn + 256, 0, 1);

    acc[0] = wmma_f16(a0, b0, acc[0]);
    acc[1] = wmma_f16(a1, b0, acc[1]);
    acc[2] = wmma_f16(a0, b1, acc[2]);
    acc[3] = wmma_f16(a1, b1, acc[3]);
    acc[4] = wmma_f16(a0, b2, acc[4]);
    acc[5] = wmma_f16(a1, b2, acc[5]);
    acc[6] = wmma_f16(a0, b3, acc[6]);
    acc[7] = wmma_f16(a1, b3, acc[7]);

    a0 = na0; a1 = na1; b0 = nb0; b1 = nb1; b2 = nb2; b3 = nb3;
  }
}

// ------------------------------------------------- QKV projection: y = x W^T
template <int MODE>
__global__ __launch_bounds__(128) void gemm_qkv(
    const _Float16* __restrict__ A, const _Float16* __restrict__ W,
    _Float16* __restrict__ out) {
  constexpr int K = D_MODEL;
  const int wave = threadIdx.x >> 5;
  const int lane = threadIdx.x & 31;
  const int lh   = lane & 15;
  const int hi   = (lane >> 4) & 1;
  const int m0   = blockIdx.x * 128 + wave * 32;
  const int n0   = blockIdx.y * 64;

  const _Float16* ap0 = A + (size_t)(m0 + lh) * K + hi * 16;
  const _Float16* ap1 = ap0 + (size_t)16 * K;
  const _Float16* bp  = W + (size_t)(n0 + lh) * K + hi * 16;

  v8f acc[8] = {};
  gemm_core(ap0, ap1, bp, acc);

#pragma unroll
  for (int j = 0; j < 4; ++j) {
    const int n = n0 + j * 16 + lh;
    const int h = n >> 6, dh = n & 63;
#pragma unroll
    for (int g = 0; g < 2; ++g) {
#pragma unroll
      for (int i = 0; i < 8; ++i) {
        const int m = m0 + g * 16 + hi * 8 + i;
        const int bidx = m >> 11, t = m & (TT - 1);
        const float v = acc[j * 2 + g][i];
        if (MODE == 0) {
          out[(((size_t)bidx * NUM_HEADS + h) * TT + t) * HEAD_DIM + dh] =
              (_Float16)(v * 0.125f);
        } else if (MODE == 1) {
          out[(((size_t)bidx * NUM_HEADS + h) * TT + t) * HEAD_DIM + dh] = (_Float16)v;
        } else {
          out[(((size_t)bidx * NUM_HEADS + h) * HEAD_DIM + dh) * TT + t] = (_Float16)v;
        }
      }
    }
  }
}

// ----------------------------------------- out projection: y = A Wo^T + bo
__global__ __launch_bounds__(128) void gemm_out(
    const _Float16* __restrict__ A, const _Float16* __restrict__ W,
    const float* __restrict__ bias, float* __restrict__ out) {
  constexpr int K = D_MODEL;
  const int wave = threadIdx.x >> 5;
  const int lane = threadIdx.x & 31;
  const int lh   = lane & 15;
  const int hi   = (lane >> 4) & 1;
  const int m0   = blockIdx.x * 128 + wave * 32;
  const int n0   = blockIdx.y * 64;

  const _Float16* ap0 = A + (size_t)(m0 + lh) * K + hi * 16;
  const _Float16* ap1 = ap0 + (size_t)16 * K;
  const _Float16* bp  = W + (size_t)(n0 + lh) * K + hi * 16;

  v8f acc[8] = {};
  gemm_core(ap0, ap1, bp, acc);

#pragma unroll
  for (int j = 0; j < 4; ++j) {
    const int n = n0 + j * 16 + lh;
    const float bn = bias[n];
#pragma unroll
    for (int g = 0; g < 2; ++g) {
#pragma unroll
      for (int i = 0; i < 8; ++i) {
        const int m = m0 + g * 16 + hi * 8 + i;
        out[(size_t)m * D_MODEL + n] = acc[j * 2 + g][i] + bn;
      }
    }
  }
}

// ---------------------------------------------- flash attention with ALiBi
// Qh,Kh: (B,H,T,Dh) f16 (Q pre-scaled).  Vt: (B,H,Dh,T) f16.
// K/V tiles are staged into double-buffered LDS by the Tensor Data Mover
// (wave 0 issues, TENSORcnt-tracked), all 4 waves consume WMMA fragments
// from LDS.  attn out: (B,T,H*Dh) f16.
__global__ __launch_bounds__(128) void attn_alibi(
    const _Float16* __restrict__ Qh, const _Float16* __restrict__ Kh,
    const _Float16* __restrict__ Vt, _Float16* __restrict__ attn) {
  const int wave = threadIdx.x >> 5;
  const int lane = threadIdx.x & 31;
  const int lh   = lane & 15;
  const int hi   = (lane >> 4) & 1;
  const int bh   = blockIdx.y;
  const int b    = bh >> 4, h = bh & 15;
  const int qbase = blockIdx.x * 64 + wave * 16;
  const float slope = exp2f(-0.5f * (float)(h + 1));

  const _Float16* kbp = Kh + (size_t)bh * TT * HEAD_DIM;
  const _Float16* vbp = Vt + (size_t)bh * HEAD_DIM * TT;

  __shared__ alignas(32) _Float16 ktile[2][32 * KROW];  // keys x dh (padded)
  __shared__ alignas(32) _Float16 vtile[2][64 * VROW];  // dh x keys (padded)
  __shared__ alignas(32) _Float16 pbuf[4][16 * 32];     // per-wave P tile

  const _Float16* qp = Qh + ((size_t)bh * TT + (qbase + lh)) * HEAD_DIM + hi * 16;
  v16h qa0 = cat(ld8(qp),      ld8(qp + 8));    // dh 0..31
  v16h qa1 = cat(ld8(qp + 32), ld8(qp + 40));   // dh 32..63

  v8f o[4] = {};
  float mrow[8], lrow[8];
#pragma unroll
  for (int i = 0; i < 8; ++i) { mrow[i] = -__builtin_inff(); lrow[i] = 0.f; }

  const int rowb = qbase + hi * 8;
  const int bkb  = hi * 16;

  auto stage = [&](int st, int ktn) {
#if HAVE_TDM
    if (wave == 0) {
      tdm_load_2d((unsigned)(uintptr_t)&ktile[st][0], kbp + (size_t)ktn * HEAD_DIM,
                  TDM_W0_K, /*dim0=*/HEAD_DIM, /*tile0=*/HEAD_DIM, /*tile1=*/32,
                  /*stride0=*/HEAD_DIM);
      tdm_load_2d((unsigned)(uintptr_t)&vtile[st][0], vbp + ktn,
                  TDM_W0_V, /*dim0=*/TT, /*tile0=*/32, /*tile1=*/HEAD_DIM,
                  /*stride0=*/TT);
      // the two ops just issued stay in flight; everything older is drained,
      // so the *other* stage is complete after this wait.
      __builtin_amdgcn_s_wait_tensorcnt(2);
    }
#else
    {  // cooperative fallback: 128 threads x 16 halfs per tile
      const int idx = (int)threadIdx.x * 16;
      {
        const int row = idx >> 6, col = idx & 63;
        const _Float16* src = kbp + (size_t)(ktn + row) * HEAD_DIM + col;
        _Float16* dst = &ktile[st][row * KROW + col];
        *(v8h*)dst = ld8(src); *(v8h*)(dst + 8) = ld8(src + 8);
      }
      {
        const int row = idx >> 5, col = idx & 31;
        const _Float16* src = vbp + (size_t)row * TT + ktn + col;
        _Float16* dst = &vtile[st][row * VROW + col];
        *(v8h*)dst = ld8(src); *(v8h*)(dst + 8) = ld8(src + 8);
      }
    }
#endif
  };

  stage(0, 0);  // prologue fill of stage 0

#pragma unroll 1
  for (int kt = 0; kt < TT; kt += 32) {
    const int st = (kt >> 5) & 1;
    __syncthreads();                        // B0: prior readers of st^1 done
    stage(st ^ 1, (kt + 32) & (TT - 1));    // async refill of other stage
    __syncthreads();                        // B1: stage st valid for all waves

    // ---- S = Q K^T from LDS K tile
    const _Float16* kl = &ktile[st][0];
    v16h b00, b01, b10, b11;
    {
      const _Float16* kr0 = kl + lh * KROW + bkb;
      const _Float16* kr1 = kl + (16 + lh) * KROW + bkb;
      b00 = cat(ld8(kr0),      ld8(kr0 + 8));
      b01 = cat(ld8(kr0 + 32), ld8(kr0 + 40));
      b10 = cat(ld8(kr1),      ld8(kr1 + 8));
      b11 = cat(ld8(kr1 + 32), ld8(kr1 + 40));
    }
    v8f s0 = {}, s1 = {};
    s0 = wmma_f16(qa0, b00, s0);
    s0 = wmma_f16(qa1, b01, s0);
    s1 = wmma_f16(qa0, b10, s1);
    s1 = wmma_f16(qa1, b11, s1);

    // ---- ALiBi bias + online softmax
    const int k0 = kt + lh, k1 = kt + 16 + lh;
#pragma unroll
    for (int i = 0; i < 8; ++i) {
      const int r = rowb + i;
      float x0 = s0[i] - slope * fabsf((float)(r - k0));
      float x1 = s1[i] - slope * fabsf((float)(r - k1));
      float mx = fmaxf(x0, x1);
      mx = fmaxf(mx, __shfl_xor(mx, 1, 32));
      mx = fmaxf(mx, __shfl_xor(mx, 2, 32));
      mx = fmaxf(mx, __shfl_xor(mx, 4, 32));
      mx = fmaxf(mx, __shfl_xor(mx, 8, 32));
      const float mn    = fmaxf(mrow[i], mx);
      const float alpha = __expf(mrow[i] - mn);
      mrow[i] = mn;
      const float p0 = __expf(x0 - mn);
      const float p1 = __expf(x1 - mn);
      float rs = p0 + p1;
      rs += __shfl_xor(rs, 1, 32);
      rs += __shfl_xor(rs, 2, 32);
      rs += __shfl_xor(rs, 4, 32);
      rs += __shfl_xor(rs, 8, 32);
      lrow[i] = lrow[i] * alpha + rs;
      o[0][i] *= alpha; o[1][i] *= alpha; o[2][i] *= alpha; o[3][i] *= alpha;
      pbuf[wave][(i + hi * 8) * 32 + lh]      = (_Float16)p0;
      pbuf[wave][(i + hi * 8) * 32 + lh + 16] = (_Float16)p1;
    }
    __syncthreads();                        // B2: P tile visible

    // ---- O += P V from LDS V tile
    const _Float16* pp = &pbuf[wave][lh * 32 + hi * 16];
    v16h pa = cat(ld8(pp), ld8(pp + 8));
    const _Float16* vl = &vtile[st][0];
    {
      const _Float16* vr0 = vl + (0 * 16 + lh) * VROW + bkb;
      const _Float16* vr1 = vl + (1 * 16 + lh) * VROW + bkb;
      const _Float16* vr2 = vl + (2 * 16 + lh) * VROW + bkb;
      const _Float16* vr3 = vl + (3 * 16 + lh) * VROW + bkb;
      o[0] = wmma_f16(pa, cat(ld8(vr0), ld8(vr0 + 8)), o[0]);
      o[1] = wmma_f16(pa, cat(ld8(vr1), ld8(vr1 + 8)), o[1]);
      o[2] = wmma_f16(pa, cat(ld8(vr2), ld8(vr2 + 8)), o[2]);
      o[3] = wmma_f16(pa, cat(ld8(vr3), ld8(vr3 + 8)), o[3]);
    }
  }

  // ---- normalize and store (B,T,H*Dh) f16 for the output projection
#pragma unroll
  for (int j = 0; j < 4; ++j) {
    const int col = h * HEAD_DIM + j * 16 + lh;
#pragma unroll
    for (int i = 0; i < 8; ++i) {
      const int t = rowb + i;
      attn[((size_t)b * TT + t) * D_MODEL + col] = (_Float16)(o[j][i] / lrow[i]);
    }
  }
}

// ---------------------------------------------------------------------------
extern "C" void kernel_launch(void* const* d_in, const int* in_sizes, int n_in,
                              void* d_out, int out_size, void* d_ws, size_t ws_size,
                              hipStream_t stream) {
  (void)in_sizes; (void)n_in; (void)out_size; (void)ws_size;
  const float* x  = (const float*)d_in[0];
  const float* wq = (const float*)d_in[1];
  const float* wk = (const float*)d_in[2];
  const float* wv = (const float*)d_in[3];
  const float* wo = (const float*)d_in[4];
  const float* bo = (const float*)d_in[5];

  char* ws = (char*)d_ws;
  size_t off = 0;
  auto take = [&](size_t bytes) -> char* {
    char* p = ws + off;
    off = (off + bytes + 255) & ~(size_t)255;
    return p;
  };

  const int XE = MROWS * D_MODEL;
  const int WE = D_MODEL * D_MODEL;

  _Float16* xh  = (_Float16*)take((size_t)XE * 2);
  _Float16* wqh = (_Float16*)take((size_t)WE * 2);
  _Float16* wkh = (_Float16*)take((size_t)WE * 2);
  _Float16* wvh = (_Float16*)take((size_t)WE * 2);
  _Float16* woh = (_Float16*)take((size_t)WE * 2);
  _Float16* qh  = (_Float16*)take((size_t)XE * 2);  // (B,H,T,Dh)
  _Float16* kh  = (_Float16*)take((size_t)XE * 2);  // (B,H,T,Dh)
  _Float16* vt  = (_Float16*)take((size_t)XE * 2);  // (B,H,Dh,T)
  _Float16* ah  = (_Float16*)take((size_t)XE * 2);  // (B,T,D)

  cvt_f32_to_f16<<<dim3((XE / 4 + 255) / 256), dim3(256), 0, stream>>>(x,  xh,  XE);
  cvt_f32_to_f16<<<dim3((WE / 4 + 255) / 256), dim3(256), 0, stream>>>(wq, wqh, WE);
  cvt_f32_to_f16<<<dim3((WE / 4 + 255) / 256), dim3(256), 0, stream>>>(wk, wkh, WE);
  cvt_f32_to_f16<<<dim3((WE / 4 + 255) / 256), dim3(256), 0, stream>>>(wv, wvh, WE);
  cvt_f32_to_f16<<<dim3((WE / 4 + 255) / 256), dim3(256), 0, stream>>>(wo, woh, WE);

  dim3 gg(MROWS / 128, D_MODEL / 64);
  gemm_qkv<0><<<gg, dim3(128), 0, stream>>>(xh, wqh, qh);
  gemm_qkv<1><<<gg, dim3(128), 0, stream>>>(xh, wkh, kh);
  gemm_qkv<2><<<gg, dim3(128), 0, stream>>>(xh, wvh, vt);

  attn_alibi<<<dim3(TT / 64, BB * NUM_HEADS), dim3(128), 0, stream>>>(qh, kh, vt, ah);

  gemm_out<<<gg, dim3(128), 0, stream>>>(ah, woh, bo, (float*)d_out);
}
